// LlamaCls__69664369541814
// MI455X (gfx1250) — matrix operand order repaired and checked by reading, not compiled, not run on hardware
//
#include <hip/hip_runtime.h>
#include <math.h>

// Problem constants (match reference)
#define SEQ_LEN 2048
#define BATCH   2
#define HIDDEN  4096
#define VOCAB   32000
#define TOKENS  (SEQ_LEN * BATCH)   // 4096 ; A row t = s*BATCH + b (row-major [s,b,h])

// Tiling
#define BM 128                      // tokens per block
#define BN 128                      // vocab per block
#define BK 32                       // K-step (WMMA K for bf16)
#define PAD 36                      // f32 staging row stride (144B -> conflict-free)
#define BKP 40                      // bf16 tile row stride (80B -> conflict-free)
#define NTB (TOKENS / BM)           // 32 token tiles
#define NVB (VOCAB / BN)            // 250 vocab tiles

typedef __attribute__((ext_vector_type(16))) __bf16 v16bf;
typedef __attribute__((ext_vector_type(8)))  __bf16 v8bf;
typedef __attribute__((ext_vector_type(8)))  float  v8f;

// Reductions across the 16-lane half-groups of a wave32 (masks 1,2,4,8 stay
// within lanes [0,15] and [16,31], which hold disjoint C-matrix rows).
__device__ __forceinline__ float red16_max(float v) {
#pragma unroll
  for (int m = 1; m <= 8; m <<= 1) v = fmaxf(v, __shfl_xor(v, m, 32));
  return v;
}
__device__ __forceinline__ float red16_sum(float v) {
#pragma unroll
  for (int m = 1; m <= 8; m <<= 1) v += __shfl_xor(v, m, 32);
  return v;
}

__device__ __forceinline__ v8bf cvt8(float4 a, float4 b) {
  v8bf t;
  t[0]=(__bf16)a.x; t[1]=(__bf16)a.y; t[2]=(__bf16)a.z; t[3]=(__bf16)a.w;
  t[4]=(__bf16)b.x; t[5]=(__bf16)b.y; t[6]=(__bf16)b.z; t[7]=(__bf16)b.w;
  return t;
}

// Async DMA of one BMxBK f32 tile (global, row stride HIDDEN) into LDS
// (row stride PAD), 16B per lane-op, 4 wave-instructions per thread.
// Tracked by ASYNCcnt (cdna5 §10.7/§15.18.3), not LOADcnt.
__device__ __forceinline__ void async_tile(unsigned ldsBase, const float* gbase, int k0) {
  const int tid = threadIdx.x;
#pragma unroll
  for (int t = 0; t < 4; ++t) {
    const int chunk = tid + t * 256;          // 0..1023 16B-chunks of the tile
    const int row   = chunk >> 3;             // 0..127
    const int c16   = chunk & 7;              // 16B column within the 128B row
    const float* g  = gbase + (size_t)row * HIDDEN + k0 + c16 * 4;
    const unsigned l = ldsBase + (unsigned)(row * PAD + c16 * 4) * 4u;
    asm volatile("global_load_async_to_lds_b128 %0, %1, off"
                 :: "v"(l), "v"(g) : "memory");
  }
}

// Kernel 1: async double-buffered bf16 WMMA GEMM (A x W^T) fused with
// online-softmax partials. Grid (NTB, NVB), blockIdx.x fastest: the 32 blocks
// of one vocab tile are co-resident, so each 2MB W tile is an L2 hit after the
// first reader and the 67MB A matrix stays resident in the 192MB L2.
__global__ __launch_bounds__(256)
void lmhead_ce_partial(const float* __restrict__ A, const float* __restrict__ W,
                       const int* __restrict__ labels,
                       float* __restrict__ wsM, float* __restrict__ wsS,
                       float* __restrict__ wsT)
{
  __shared__ __align__(128) float  shAf[2][BM * PAD];  // f32 async staging (double)
  __shared__ __align__(128) float  shBf[2][BN * PAD];
  __shared__ __align__(32)  __bf16 sbA[BM * BKP];      // bf16 compute tiles (single)
  __shared__ __align__(32)  __bf16 sbB[BN * BKP];
  __shared__ float smM[2][BM], smS[2][BM], smT[2][BM];
  __shared__ int   labS[BM];

  const int tid  = threadIdx.x;
  const int wave = tid >> 5;
  const int lane = tid & 31;
  const int n    = lane & 15;      // fragment row/col index within 16
  const int half = lane >> 4;      // lane half-group
  const int mt   = wave & 3;       // wave's 32-token quadrant
  const int vc   = wave >> 2;      // wave's 64-vocab half

  const int tokBase = blockIdx.x * BM;
  const int vBase   = blockIdx.y * BN;

  if (tid < BM) {
    const int t = tokBase + tid;                       // t = s*BATCH + b
    labS[tid] = labels[(t % BATCH) * SEQ_LEN + (t / BATCH)];
  }

  const float* Abase = A + (size_t)tokBase * HIDDEN;
  const float* Wbase = W + (size_t)vBase   * HIDDEN;

  // LDS byte offsets of the staging buffers (flat addr low 32 bits == LDS
  // offset; the async VDST operand is a wave-relative LDS byte address).
  const unsigned offA0 = (unsigned)(size_t)(void*)&shAf[0][0];
  const unsigned offA1 = (unsigned)(size_t)(void*)&shAf[1][0];
  const unsigned offB0 = (unsigned)(size_t)(void*)&shBf[0][0];
  const unsigned offB1 = (unsigned)(size_t)(void*)&shBf[1][0];

  v8f acc[2][4] = {};   // 32 tokens x 64 vocab per wave (8 WMMA C tiles)

  // cooperative-convert coords: thread -> (row, 16-float chunk)
  const int crow = tid >> 1;
  const int ckof = (tid & 1) * 16;

  // prologue: start DMA of K-slab 0 into staging buffer 0
  async_tile(offA0, Abase, 0);
  async_tile(offB0, Wbase, 0);

  int buf = 0;
  for (int k0 = 0; k0 < HIDDEN; k0 += BK) {
    const int kn = k0 + BK;
    asm volatile("s_wait_asynccnt 0x0" ::: "memory");  // slab k landed in buf
    __syncthreads();  // copies visible to all; prev iter's bf16 reads finished

    if (kn < HIDDEN) {
      // stream next slab into the other staging buffer while we compute
      async_tile(buf ? offA0 : offA1, Abase, kn);
      async_tile(buf ? offB0 : offB1, Wbase, kn);
      if (kn + BK < HIDDEN)   // warm L2 two slabs ahead for the streaming matrix
        __builtin_prefetch((const void*)(Wbase + (size_t)crow * HIDDEN +
                                         ckof + kn + BK), 0, 1);
    }

    // ---- cooperative tile convert: f32 staging -> bf16 tiles (once, not per wave)
    {
      const float4* p = (const float4*)(&shAf[buf][crow * PAD + ckof]);
      float4 a0 = p[0], a1 = p[1], a2 = p[2], a3 = p[3];
      const float4* q = (const float4*)(&shBf[buf][crow * PAD + ckof]);
      float4 b0 = q[0], b1 = q[1], b2 = q[2], b3 = q[3];
      v8bf* da = (v8bf*)&sbA[crow * BKP + ckof];
      da[0] = cvt8(a0, a1); da[1] = cvt8(a2, a3);
      v8bf* db = (v8bf*)&sbB[crow * BKP + ckof];
      db[0] = cvt8(b0, b1); db[1] = cvt8(b2, b3);
    }
    __syncthreads();  // bf16 tiles complete

    // ---- A fragments (ISA 16-bit A 16x32 layout): lane holds M = n;
    // elems 0..7 -> K = half*8+0..7, elems 8..15 -> K = 16+half*8+0..7
    v16bf af[2];
#pragma unroll
    for (int i = 0; i < 2; ++i) {
      const __bf16* base = &sbA[(mt * 32 + i * 16 + n) * BKP];
      v8bf lo = *(const v8bf*)(base + half * 8);
      v8bf hi = *(const v8bf*)(base + 16 + half * 8);
#pragma unroll
      for (int e = 0; e < 8; ++e) { af[i][e] = lo[e]; af[i][e + 8] = hi[e]; }
    }
    // ---- B fragments (16-bit B 32x16): lane holds N = n;
    // elems 0..7 -> K = half*8+0..7, elems 8..15 -> K = 16+half*8+0..7
#pragma unroll
    for (int j = 0; j < 4; ++j) {
      const __bf16* base = &sbB[(vc * 64 + j * 16 + n) * BKP];
      v8bf lo = *(const v8bf*)(base + half * 8);
      v8bf hi = *(const v8bf*)(base + 16 + half * 8);
      v16bf bfv;
#pragma unroll
      for (int e = 0; e < 8; ++e) { bfv[e] = lo[e]; bfv[e + 8] = hi[e]; }
#pragma unroll
      for (int i = 0; i < 2; ++i)
        acc[i][j] = __builtin_amdgcn_wmma_f32_16x16x32_bf16(
            false, af[i], false, bfv, (short)0, acc[i][j], false, false);
    }
    buf ^= 1;
  }

  // ---- fused epilogue: per-token partial (max, sum-exp, target logit) over
  // this block's 128 vocab columns. C layout: VGPR r holds M = half*8 + r, N = n.
#pragma unroll
  for (int i = 0; i < 2; ++i) {
#pragma unroll
    for (int r = 0; r < 8; ++r) {
      const float v0 = acc[i][0][r], v1 = acc[i][1][r];
      const float v2 = acc[i][2][r], v3 = acc[i][3][r];
      const int M   = half * 8 + r;
      const int li  = mt * 32 + i * 16 + M;          // token row within block
      const int lab = labS[li];
      const int cl  = lab - (vBase + vc * 64);       // col within this wave's 64
      float tv = -INFINITY;
      if (cl >= 0 && cl < 64 && n == (cl & 15)) {
        const int j = cl >> 4;
        tv = (j == 0) ? v0 : (j == 1) ? v1 : (j == 2) ? v2 : v3;
      }
      float mx = fmaxf(fmaxf(v0, v1), fmaxf(v2, v3));
      mx = red16_max(mx);
      tv = red16_max(tv);     // exactly one finite contributor (if label in range)
      float s = __expf(v0 - mx) + __expf(v1 - mx) + __expf(v2 - mx) + __expf(v3 - mx);
      s = red16_sum(s);
      if (n == 0) { smM[vc][li] = mx; smS[vc][li] = s; smT[vc][li] = tv; }
    }
  }
  __syncthreads();

  if (tid < BM) {   // merge the two vocab halves, emit one partial per (token, vocab-tile)
    const float m0 = smM[0][tid], m1 = smM[1][tid];
    const float mm = fmaxf(m0, m1);
    const float ss = smS[0][tid] * __expf(m0 - mm) + smS[1][tid] * __expf(m1 - mm);
    const float tt = fmaxf(smT[0][tid], smT[1][tid]);
    const size_t idx = (size_t)(tokBase + tid) * NVB + blockIdx.y;
    wsM[idx] = mm; wsS[idx] = ss; wsT[idx] = tt;
  }
}

// Kernel 2: merge 250 vocab-tile partials per token -> nll (online logsumexp).
__global__ __launch_bounds__(256)
void ce_reduce(const float* __restrict__ wsM, const float* __restrict__ wsS,
               const float* __restrict__ wsT, float* __restrict__ out)
{
  const int t = blockIdx.x * blockDim.x + threadIdx.x;
  if (t >= TOKENS) return;
  const float* pm = wsM + (size_t)t * NVB;
  const float* ps = wsS + (size_t)t * NVB;
  const float* pt = wsT + (size_t)t * NVB;
  float M = -INFINITY, S = 0.f, T = -INFINITY;
  for (int i = 0; i < NVB; ++i) {
    const float m  = pm[i];
    const float nM = fmaxf(M, m);
    S = S * __expf(M - nM) + ps[i] * __expf(m - nM);
    M = nM;
    T = fmaxf(T, pt[i]);
  }
  // nll = logsumexp - target_logit ; output layout [b, s] with t = s*BATCH + b
  out[(t % BATCH) * SEQ_LEN + (t / BATCH)] = (M + __logf(S)) - T;
}

extern "C" void kernel_launch(void* const* d_in, const int* in_sizes, int n_in,
                              void* d_out, int out_size, void* d_ws, size_t ws_size,
                              hipStream_t stream)
{
  (void)in_sizes; (void)n_in; (void)out_size; (void)ws_size;
  const float* A = (const float*)d_in[0];   // hidden_states [2048,2,4096] f32
  const float* W = (const float*)d_in[1];   // weight [32000,4096] f32
  const int*   L = (const int*)d_in[2];     // labels [2,2048] (int)
  float* out = (float*)d_out;               // nll [2,2048] f32

  float* wsM = (float*)d_ws;                            // 4096*250 f32
  float* wsS = wsM + (size_t)TOKENS * NVB;
  float* wsT = wsS + (size_t)TOKENS * NVB;              // total ~12.3 MB

  dim3 grid(NTB, NVB);
  lmhead_ce_partial<<<grid, dim3(256), 0, stream>>>(A, W, L, wsM, wsS, wsT);
  ce_reduce<<<dim3((TOKENS + 255) / 256), dim3(256), 0, stream>>>(wsM, wsS, wsT, out);
}
